// AttHeteroRGCNLayer_26310969655540
// MI455X (gfx1250) — matrix operand order repaired and checked by reading, not compiled, not run on hardware
//
#include <hip/hip_runtime.h>

// ---------------------------------------------------------------------------
// AttHeteroRGCN GAT layer for gfx1250 (MI455X).
//   Wh_t  = x_drug @ W_t  + b_t           (WMMA bf16x3, fp32 accumulate)
//   Wh_rt = x_dis  @ W_rt + b_rt
//   per-etype: segment softmax of leaky_relu(sSrc[src]+sDst[dst]) over dst,
//              h_dst = segsum(alpha * Wh_src[src])
// Output d_out = [h_drug (50000x256) | h_dis (50000x256)] fp32.
// ---------------------------------------------------------------------------

#define N_DRUG 50000
#define N_DIS  50000
#define N_EDGE 500000
#define DIM    256
#define NSTRIP 8            // N-tiles per wave in the GEMM (16x128 strip)

typedef __bf16 bf16_t;
typedef bf16_t          v16bf  __attribute__((ext_vector_type(16)));
typedef unsigned short  v16u16 __attribute__((ext_vector_type(16)));
typedef unsigned int    v8u32  __attribute__((ext_vector_type(8)));
typedef float           v8f    __attribute__((ext_vector_type(8)));

// ---- monotone float<->uint key for atomicMax on floats --------------------
static __device__ __forceinline__ unsigned fkey(float f) {
    unsigned u = __float_as_uint(f);
    return (u & 0x80000000u) ? ~u : (u | 0x80000000u);
}
static __device__ __forceinline__ float keyf(unsigned k) {
    unsigned u = (k & 0x80000000u) ? (k & 0x7fffffffu) : ~k;
    return __uint_as_float(u);
}

// ---------------------------------------------------------------------------
// zero fill (float4 granularity, grid-stride)
// ---------------------------------------------------------------------------
__global__ void k_zero_f4(float4* __restrict__ p, long long n4) {
    long long i = (long long)blockIdx.x * blockDim.x + threadIdx.x;
    long long stride = (long long)gridDim.x * blockDim.x;
    float4 z; z.x = z.y = z.z = z.w = 0.f;
    for (; i < n4; i += stride) p[i] = z;
}

// ---------------------------------------------------------------------------
// One-time weight prep: W[256,256] (K-major) -> transposed bf16 hi/lo splits
// WtHi/WtLo[n*256 + k], so a B fragment (K=h*16..h*16+15 at column n) is one
// contiguous 32-byte load.  f ~= hi + lo (truncation split).
// ---------------------------------------------------------------------------
__global__ void k_wsplit(const float* __restrict__ W,
                         unsigned short* __restrict__ WtHi,
                         unsigned short* __restrict__ WtLo) {
    const int idx = blockIdx.x * blockDim.x + threadIdx.x;   // 65536
    const int k = idx >> 8, n = idx & 255;
    const float f = W[k * DIM + n];
    const unsigned u = __float_as_uint(f);
    const float lo = f - __uint_as_float(u & 0xFFFF0000u);
    WtHi[n * DIM + k] = (unsigned short)(u >> 16);
    WtLo[n * DIM + k] = (unsigned short)(__float_as_uint(lo) >> 16);
}

// ---------------------------------------------------------------------------
// GEMM: Y[M,256] = X[M,256] @ W[256,256] + b  via v_wmma_f32_16x16x32_bf16.
// bf16x3 split for ~fp32 accuracy.  One wave computes a 16x(16*NSTRIP) strip:
// A fragment converted once per K-step, reused across NSTRIP tiles;
// B fragments are preconverted contiguous bf16 loads (zero VALU).
// blockDim = 256 (8 waves).
// ---------------------------------------------------------------------------
__global__ void k_gemm_wmma(const float* __restrict__ X,
                            const unsigned short* __restrict__ WtHi,
                            const unsigned short* __restrict__ WtLo,
                            const float* __restrict__ b,
                            float* __restrict__ Y, int M) {
    const int wgl = blockIdx.x * 8 + (threadIdx.x >> 5);  // global wave id
    const int mt  = wgl >> 1;                             // M-tile index
    const int ns  = wgl & 1;                              // N half (0 or 1)
    if (mt * 16 >= M) return;                             // uniform per wave
    const int lane = threadIdx.x & 31;
    const int h    = lane >> 4;                           // half-wave
    const int ml   = lane & 15;                           // row (A) / col (B,C)
    const int m0   = mt * 16;
    const int n0b  = ns * (NSTRIP * 16);
    const float* __restrict__ xrow = X + (long long)(m0 + ml) * DIM;

    v8f zero = {};
    v8f acc[NSTRIP];
    #pragma unroll
    for (int t = 0; t < NSTRIP; ++t) acc[t] = zero;

    for (int k0 = 0; k0 < DIM; k0 += 32) {
        if (k0 + 64 < DIM)                                // stream A ahead
            __builtin_prefetch(xrow + k0 + 64, 0, 1);

        // --- A fragment (contiguous K order, matching B): per lane the needed K
        // values are one contiguous 16-float run at k0+16h.
        const float4 c0 = *(const float4*)(xrow + k0 + 16 * h);
        const float4 c1 = *(const float4*)(xrow + k0 + 16 * h + 4);
        const float4 c2 = *(const float4*)(xrow + k0 + 16 * h + 8);
        const float4 c3 = *(const float4*)(xrow + k0 + 16 * h + 12);
        const float af[16] = {c0.x, c0.y, c0.z, c0.w, c1.x, c1.y, c1.z, c1.w,
                              c2.x, c2.y, c2.z, c2.w, c3.x, c3.y, c3.z, c3.w};
        v8u32 ahp, alp;
        #pragma unroll
        for (int p = 0; p < 8; ++p) {
            const unsigned u0 = __float_as_uint(af[2 * p]);
            const unsigned u1 = __float_as_uint(af[2 * p + 1]);
            ahp[p] = (u1 & 0xFFFF0000u) | (u0 >> 16);
            const float l0 = af[2 * p]     - __uint_as_float(u0 & 0xFFFF0000u);
            const float l1 = af[2 * p + 1] - __uint_as_float(u1 & 0xFFFF0000u);
            alp[p] = (__float_as_uint(l1) & 0xFFFF0000u) |
                     (__float_as_uint(l0) >> 16);
        }
        const v16bf Ah = __builtin_bit_cast(v16bf, ahp);
        const v16bf Al = __builtin_bit_cast(v16bf, alp);

        #pragma unroll
        for (int t = 0; t < NSTRIP; ++t) {
            const int n0 = n0b + t * 16;
            const long long boff = (long long)(n0 + ml) * DIM + k0 + h * 16;
            const v16bf Bh = __builtin_bit_cast(v16bf, *(const v16u16*)(WtHi + boff));
            const v16bf Bl = __builtin_bit_cast(v16bf, *(const v16u16*)(WtLo + boff));
            acc[t] = __builtin_amdgcn_wmma_f32_16x16x32_bf16(false, Ah, false, Bh,
                                                             (short)0, acc[t], false, false);
            acc[t] = __builtin_amdgcn_wmma_f32_16x16x32_bf16(false, Ah, false, Bl,
                                                             (short)0, acc[t], false, false);
            acc[t] = __builtin_amdgcn_wmma_f32_16x16x32_bf16(false, Al, false, Bh,
                                                             (short)0, acc[t], false, false);
        }
    }
    // C/D layout: VGPR r, lane: M = r + 8h, N = ml
    #pragma unroll
    for (int t = 0; t < NSTRIP; ++t) {
        const int n0 = n0b + t * 16;
        const float bias = b[n0 + ml];
        #pragma unroll
        for (int r = 0; r < 8; ++r)
            Y[(long long)(m0 + r + 8 * h) * DIM + n0 + ml] = acc[t][r] + bias;
    }
}

// ---------------------------------------------------------------------------
// node scores: sA[row] = Wh[row]·aA ; sB[row] = Wh[row]·aB   (wave per row)
// ---------------------------------------------------------------------------
__global__ void k_scores(const float* __restrict__ Wh,
                         const float* __restrict__ aA,
                         const float* __restrict__ aB,
                         float* __restrict__ sA, float* __restrict__ sB,
                         int nrows) {
    const int lane = threadIdx.x & 31;
    const int row  = blockIdx.x * 8 + (threadIdx.x >> 5);
    if (row >= nrows) return;
    const float* __restrict__ w = Wh + (long long)row * DIM;
    float s1 = 0.f, s2 = 0.f;
    #pragma unroll
    for (int j = 0; j < 8; ++j) {
        const float v = w[j * 32 + lane];
        s1 += v * aA[j * 32 + lane];
        s2 += v * aB[j * 32 + lane];
    }
    #pragma unroll
    for (int off = 16; off >= 1; off >>= 1) {
        s1 += __shfl_xor(s1, off, 32);
        s2 += __shfl_xor(s2, off, 32);
    }
    if (lane == 0) { sA[row] = s1; sB[row] = s2; }
}

// ---------------------------------------------------------------------------
// pass 1: per-edge logit + segment max (ordered-uint atomicMax)
// ---------------------------------------------------------------------------
__global__ void k_edge_logit_max(const float* __restrict__ sSrc,
                                 const float* __restrict__ sDst,
                                 const int* __restrict__ src,
                                 const int* __restrict__ dst,
                                 float* __restrict__ elog,
                                 unsigned* __restrict__ mkey, int E) {
    const int e = blockIdx.x * blockDim.x + threadIdx.x;
    if (e >= E) return;
    float l = sSrc[src[e]] + sDst[dst[e]];
    l = (l >= 0.f) ? l : 0.01f * l;                 // leaky_relu(0.01)
    elog[e] = l;
    atomicMax(&mkey[dst[e]], fkey(l));
}

// ---------------------------------------------------------------------------
// pass 2: e = exp(l - m[dst]) ; Z[dst] += e   (elog overwritten with e)
// ---------------------------------------------------------------------------
__global__ void k_edge_exp_sum(const int* __restrict__ dst,
                               float* __restrict__ elog,
                               const unsigned* __restrict__ mkey,
                               float* __restrict__ z, int E) {
    const int e = blockIdx.x * blockDim.x + threadIdx.x;
    if (e >= E) return;
    const int d = dst[e];
    const float w = __expf(elog[e] - keyf(mkey[d]));
    elog[e] = w;
    atomicAdd(&z[d], w);
}

// ---------------------------------------------------------------------------
// pass 3: out[dst] += (e/Z[dst]) * Wh_src[src] ; 64 threads/edge, float4 each
// ---------------------------------------------------------------------------
__global__ void k_edge_scatter(const float* __restrict__ WhSrc,
                               const int* __restrict__ src,
                               const int* __restrict__ dst,
                               const float* __restrict__ elog,
                               const float* __restrict__ z,
                               float* __restrict__ out, int E) {
    const long long t = (long long)blockIdx.x * blockDim.x + threadIdx.x;
    const int e = (int)(t >> 6);
    if (e >= E) return;
    const int c = (int)(t & 63);
    const int s = src[e], d = dst[e];
    const float w = elog[e] / z[d];
    const float4 v = ((const float4*)(WhSrc + (long long)s * DIM))[c];
    float* o = out + (long long)d * DIM + c * 4;
    atomicAdd(o + 0, w * v.x);
    atomicAdd(o + 1, w * v.y);
    atomicAdd(o + 2, w * v.z);
    atomicAdd(o + 3, w * v.w);
}

// ---------------------------------------------------------------------------
extern "C" void kernel_launch(void* const* d_in, const int* in_sizes, int n_in,
                              void* d_out, int out_size, void* d_ws, size_t ws_size,
                              hipStream_t stream) {
    const float* x_drug = (const float*)d_in[0];
    const float* x_dis  = (const float*)d_in[1];
    const float* W_t    = (const float*)d_in[2];
    const float* b_t    = (const float*)d_in[3];
    const float* W_rt   = (const float*)d_in[4];
    const float* b_rt   = (const float*)d_in[5];
    const float* a_t    = (const float*)d_in[6];
    const float* a_rt   = (const float*)d_in[7];
    const int*   src_t  = (const int*)d_in[8];
    const int*   dst_t  = (const int*)d_in[9];
    const int*   src_rt = (const int*)d_in[10];
    const int*   dst_rt = (const int*)d_in[11];
    float* out = (float*)d_out;                    // [h_drug | h_dis]
    float* ws  = (float*)d_ws;

    const long long NT = (long long)N_DRUG * DIM;  // 12.8M floats
    float*    Wh_t   = ws;                         // [N_DRUG,256]
    float*    Wh_rt  = ws + NT;                    // [N_DIS,256]
    float*    sA_t   = ws + 2 * NT;                // Wh_t  · a_t[:256]   (treats src)
    float*    sB_t   = sA_t + N_DRUG;              // Wh_t  · a_rt[256:]  (rev dst)
    float*    sA_rt  = sB_t + N_DRUG;              // Wh_rt · a_rt[:256]  (rev src)
    float*    sB_rt  = sA_rt + N_DIS;              // Wh_rt · a_t[256:]   (treats dst)
    unsigned* mkey_t = (unsigned*)(sB_rt + N_DIS); // per-dis max keys
    float*    z_t    = (float*)(mkey_t + N_DIS);
    unsigned* mkey_r = (unsigned*)(z_t + N_DIS);   // per-drug max keys
    float*    z_r    = (float*)(mkey_r + N_DRUG);
    float*    e_t    = z_r + N_DRUG;               // edge weights, treats
    float*    e_r    = e_t + N_EDGE;               // edge weights, rev
    unsigned short* WtHi_t = (unsigned short*)(e_r + N_EDGE);  // 32B-aligned
    unsigned short* WtLo_t = WtHi_t + DIM * DIM;
    unsigned short* WtHi_r = WtLo_t + DIM * DIM;
    unsigned short* WtLo_r = WtHi_r + DIM * DIM;

    // ---- zero: outputs (atomic accumulators) + max/Z stats ----------------
    k_zero_f4<<<2048, 256, 0, stream>>>((float4*)out, (2 * NT) / 4);
    k_zero_f4<<<256, 256, 0, stream>>>((float4*)mkey_t,
                                       (long long)(2 * (N_DIS + N_DRUG)) / 4);

    // ---- weight prep: transpose + bf16 hi/lo split (one-time, tiny) -------
    k_wsplit<<<DIM, 256, 0, stream>>>(W_t,  WtHi_t, WtLo_t);
    k_wsplit<<<DIM, 256, 0, stream>>>(W_rt, WtHi_r, WtLo_r);

    // ---- dense projections (WMMA) : (M/16)*2 waves, 8 waves/block ---------
    const int GB = (N_DRUG / 16 * 2 + 7) / 8;      // 782 blocks
    k_gemm_wmma<<<GB, 256, 0, stream>>>(x_drug, WtHi_t, WtLo_t, b_t, Wh_t, N_DRUG);
    k_gemm_wmma<<<GB, 256, 0, stream>>>(x_dis,  WtHi_r, WtLo_r, b_rt, Wh_rt, N_DIS);

    // ---- node scores ------------------------------------------------------
    k_scores<<<N_DRUG / 8, 256, 0, stream>>>(Wh_t,  a_t,  a_rt + DIM, sA_t,  sB_t,  N_DRUG);
    k_scores<<<N_DIS  / 8, 256, 0, stream>>>(Wh_rt, a_rt, a_t  + DIM, sA_rt, sB_rt, N_DIS);

    const int EB = (N_EDGE + 255) / 256;
    const int SB = (N_EDGE * 64) / 256;            // 125000 blocks

    // ---- etype 'treats': drug -> disease, output h_dis (out + NT) ---------
    k_edge_logit_max<<<EB, 256, 0, stream>>>(sA_t, sB_rt, src_t, dst_t, e_t, mkey_t, N_EDGE);
    k_edge_exp_sum  <<<EB, 256, 0, stream>>>(dst_t, e_t, mkey_t, z_t, N_EDGE);
    k_edge_scatter  <<<SB, 256, 0, stream>>>(Wh_t, src_t, dst_t, e_t, z_t, out + NT, N_EDGE);

    // ---- etype 'rev_treats': disease -> drug, output h_drug (out) ---------
    k_edge_logit_max<<<EB, 256, 0, stream>>>(sA_rt, sB_t, src_rt, dst_rt, e_r, mkey_r, N_EDGE);
    k_edge_exp_sum  <<<EB, 256, 0, stream>>>(dst_rt, e_r, mkey_r, z_r, N_EDGE);
    k_edge_scatter  <<<SB, 256, 0, stream>>>(Wh_rt, src_rt, dst_rt, e_r, z_r, out, N_EDGE);
}
